// RNN_28561532518804
// MI455X (gfx1250) — matrix operand order repaired and checked by reading, not compiled, not run on hardware
//
#include <hip/hip_runtime.h>
#include <hip/hip_bf16.h>
#include <math.h>

// ---------------------------------------------------------------------------
// RNN on MI455X (gfx1250, wave32, WMMA + Tensor Data Mover):
//   xh  = x  @ W_xh^T + b_xh        (bf16 WMMA GEMM, TDM->LDS staged)
//   h_t = tanh(xh_t + h_{t-1} @ W_hh^T + b_hh)   (persistent 16-WG scan)
//   out = hs @ W_hy^T + b_hy        (bf16 WMMA GEMM, TDM->LDS staged)
// ---------------------------------------------------------------------------

typedef __attribute__((ext_vector_type(16))) __bf16 v16bf;
typedef __attribute__((ext_vector_type(8)))  float  v8f;
typedef __attribute__((ext_vector_type(4)))  unsigned int u32x4;
typedef __attribute__((ext_vector_type(8)))  int  i32x8;
typedef __attribute__((ext_vector_type(4)))  int  i32x4;

#define SEQ     512
#define BATCH   64
#define DIM     1024           // INPUT == HIDDEN == OUTPUT == 1024
#define MTOT    (SEQ * BATCH)  // 32768 rows for the big GEMMs
#define SCAN_WGS 16
#define KCHUNK  64             // K elements staged per TDM chunk
#define NCHUNK  (DIM / KCHUNK) // 16

// LDS layout for the TDM GEMM (bf16 element offsets), double buffered:
//   A: 64 x 64  = 4096 elems/buf ; B: 256 x 64 = 16384 elems/buf
#define LDS_A(buf)  ((buf) * 4096)
#define LDS_B(buf)  (8192 + (buf) * 16384)
#define LDS_ELEMS   40960
#define LDS_BYTES   (LDS_ELEMS * 2)   // 80 KB (of 320 KB WGP LDS)

// -------------------- workspace layout (bytes) ------------------------------
static constexpr size_t SZ_W   = (size_t)DIM * DIM;        // weight elems
static constexpr size_t SZ_X   = (size_t)MTOT * DIM;       // activation elems
static constexpr size_t OFF_WXH = 0;                       // bf16, 2 MiB
static constexpr size_t OFF_WHH = OFF_WXH + SZ_W * 2;      // bf16, 2 MiB
static constexpr size_t OFF_WHY = OFF_WHH + SZ_W * 2;      // bf16, 2 MiB
static constexpr size_t OFF_XB  = OFF_WHY + SZ_W * 2;      // x  bf16, 64 MiB
static constexpr size_t OFF_XH  = OFF_XB  + SZ_X * 2;      // xh f32, 128 MiB
static constexpr size_t OFF_HS  = OFF_XH  + SZ_X * 4;      // hs bf16, 64 MiB
static constexpr size_t OFF_HG  = OFF_HS  + SZ_X * 2;      // h ping-pong bf16
static constexpr size_t OFF_CTR = OFF_HG  + (size_t)2 * BATCH * DIM * 2; // u32

// -------------------- fp32 -> bf16 convert ----------------------------------
__global__ void f32_to_bf16(const float* __restrict__ in,
                            __bf16* __restrict__ out, int n) {
  int i = blockIdx.x * blockDim.x + threadIdx.x;
  int stride = gridDim.x * blockDim.x;
  for (; i < n; i += stride) out[i] = (__bf16)in[i];
}

// -------------------- WMMA helpers ------------------------------------------
__device__ __forceinline__ v8f wmma_bf16(v16bf a, v16bf b, v8f c) {
  return __builtin_amdgcn_wmma_f32_16x16x32_bf16(
      /*neg_a=*/false, a, /*neg_b=*/false, b,
      /*c_mod=*/(short)0, c, /*reuse_a=*/false, /*reuse_b=*/false);
}

// A fragment (16x32 bf16): lanes 0-15 -> M, K 0..7 & 16..23; lanes 16-31 -> K
// 8..15 & 24..31.  `base` = row 0 of the tile, leading dim `ld` (elements).
__device__ __forceinline__ v16bf frag_a(const __bf16* base, int ld, int k0) {
  int lane = threadIdx.x & 31;
  int m = lane & 15, half = lane >> 4;
  const __bf16* p0 = base + (size_t)m * ld + k0 + half * 8;
  v16bf f;
  *reinterpret_cast<uint4*>(&f)       = *reinterpret_cast<const uint4*>(p0);
  *(reinterpret_cast<uint4*>(&f) + 1) = *reinterpret_cast<const uint4*>(p0 + 16);
  return f;
}

// B fragment (32x16 bf16), B[k,n] = W[n,k], W row-major (k contiguous):
// lanes 0-15 -> N, K 0..7 & 16..23; lanes 16-31 -> N, K 8..15 & 24..31.
__device__ __forceinline__ v16bf frag_b(const __bf16* wrow0, int ld, int k0) {
  int lane = threadIdx.x & 31;
  int n = lane & 15, kh = lane >> 4;
  const __bf16* p = wrow0 + (size_t)n * ld + k0 + kh * 8;
  v16bf f;
  *reinterpret_cast<uint4*>(&f)       = *reinterpret_cast<const uint4*>(p);
  *(reinterpret_cast<uint4*>(&f) + 1) = *reinterpret_cast<const uint4*>(p + 16);
  return f;
}

// Global-load variants (used by the latency-bound scan kernel) with prefetch.
__device__ __forceinline__ v16bf frag_a_g(const __bf16* base, int ld, int k0) {
  int lane = threadIdx.x & 31;
  int m = lane & 15, half = lane >> 4;
  const __bf16* p0 = base + (size_t)m * ld + k0 + half * 8;
  v16bf f;
  *reinterpret_cast<uint4*>(&f)       = *reinterpret_cast<const uint4*>(p0);
  *(reinterpret_cast<uint4*>(&f) + 1) = *reinterpret_cast<const uint4*>(p0 + 16);
  __builtin_prefetch(p0 + 128, 0, 3);   // global_prefetch_b8, near-cache
  return f;
}
__device__ __forceinline__ v16bf frag_b_g(const __bf16* wrow0, int ld, int k0) {
  int lane = threadIdx.x & 31;
  int n = lane & 15, kh = lane >> 4;
  const __bf16* p = wrow0 + (size_t)n * ld + k0 + kh * 8;
  v16bf f;
  *reinterpret_cast<uint4*>(&f)       = *reinterpret_cast<const uint4*>(p);
  *(reinterpret_cast<uint4*>(&f) + 1) = *reinterpret_cast<const uint4*>(p + 16);
  __builtin_prefetch(p + 128, 0, 3);
  return f;
}

// -------------------- Tensor Data Mover: 2-D tile -> LDS --------------------
// D# per CDNA5 ISA: group0 {count=1, lds_addr,
// global_addr, type=2}; group1 {data_size=2B, tensor_dim0/1, tile_dim0/1,
// tensor_dim0_stride}. 2-D tile => remaining groups zero (NULL-equivalent).
// clang-23 6-arg builtin form: (u32x4, i32x8, i32x4, i32x4, i32x8, i32 cpol).
__device__ __forceinline__ void tdm_load_2d(unsigned lds_byte_off,
                                            const __bf16* gsrc,
                                            unsigned rows,
                                            unsigned stride_elems) {
  unsigned long long ga = (unsigned long long)(size_t)gsrc;
  u32x4 g0;
  g0[0] = (unsigned)__builtin_amdgcn_readfirstlane(1);              // count=1
  g0[1] = (unsigned)__builtin_amdgcn_readfirstlane((int)lds_byte_off);
  g0[2] = (unsigned)__builtin_amdgcn_readfirstlane((int)(ga & 0xffffffffu));
  g0[3] = (unsigned)__builtin_amdgcn_readfirstlane(
              (int)(((ga >> 32) & 0x1ffffffu) | (2u << 30)));       // type=2
  i32x8 g1;
  g1[0] = __builtin_amdgcn_readfirstlane((int)(1u << 16));          // 2B elems
  g1[1] = __builtin_amdgcn_readfirstlane((int)((unsigned)KCHUNK << 16)); // td0 lo16
  g1[2] = __builtin_amdgcn_readfirstlane((int)(rows << 16));        // td1 lo16
  g1[3] = __builtin_amdgcn_readfirstlane((int)((unsigned)KCHUNK << 16)); // tile_dim0
  g1[4] = __builtin_amdgcn_readfirstlane((int)rows);                // tile_dim1
  g1[5] = __builtin_amdgcn_readfirstlane((int)stride_elems);        // stride0
  g1[6] = 0;
  g1[7] = 0;
  i32x4 z4 = {0, 0, 0, 0};
  i32x8 z8 = {0, 0, 0, 0, 0, 0, 0, 0};
  __builtin_amdgcn_tensor_load_to_lds(g0, g1, z4, z4, z8, 0);
}

// -------------------- big GEMM: C[M,1024] = A_bf16 @ W^T + bias -------------
// Block tile 64M x 256N, K-chunk 64, TDM double-buffered through LDS.
// 8 waves: wm = wid&1 (32 M rows), wn = wid>>1 (64 N cols); per-wave
// 2x4 = 8 accumulator tiles. Grid: (M/64, 1024/256).
__global__ void __launch_bounds__(256)
gemm_bf16_bias(const __bf16* __restrict__ A,   // [M][1024] bf16, K-major
               const __bf16* __restrict__ W,   // [1024][1024] bf16 row-major
               const float*  __restrict__ bias,// [1024]
               float* __restrict__ C, int M) {
  extern __shared__ __bf16 smem[];   // LDS_ELEMS bf16 (80 KB), dynamic
  const int K = DIM, N = DIM;
  int wid = threadIdx.x >> 5;          // wave32
  int wm = wid & 1, wn = wid >> 1;
  int mblk = blockIdx.x * 64;
  int nblk = blockIdx.y * 256;
  (void)M;

  const __bf16* Ablk = A + (size_t)mblk * K;
  const __bf16* Wblk = W + (size_t)nblk * K;
  unsigned lds0 = (unsigned)__builtin_amdgcn_groupstaticsize();

  // Prologue: stage chunk 0 into buffer 0.
  if (wid == 0) {
    tdm_load_2d(lds0 + LDS_A(0) * 2, Ablk, 64, DIM);
    tdm_load_2d(lds0 + LDS_B(0) * 2, Wblk, 256, DIM);
  }

  v8f acc[2][4] = {};
  for (int kc = 0; kc < NCHUNK; ++kc) {
    int cur = kc & 1;
    if (wid == 0) {
      if (kc + 1 < NCHUNK) {
        // Issue next chunk into the other buffer, then wait for current
        // (TENSORcnt is in-order per wave: <=2 pending => chunk kc done).
        int k0 = (kc + 1) * KCHUNK;
        tdm_load_2d(lds0 + LDS_A(cur ^ 1) * 2, Ablk + k0, 64, DIM);
        tdm_load_2d(lds0 + LDS_B(cur ^ 1) * 2, Wblk + k0, 256, DIM);
        __builtin_amdgcn_s_wait_tensorcnt(2);
      } else {
        __builtin_amdgcn_s_wait_tensorcnt(0);
      }
    }
    __syncthreads();                  // chunk kc visible to all waves

    const __bf16* Abuf = smem + LDS_A(cur) + (size_t)(wm * 32) * KCHUNK;
    const __bf16* Bbuf = smem + LDS_B(cur) + (size_t)(wn * 64) * KCHUNK;
    for (int ks = 0; ks < KCHUNK; ks += 32) {
      v16bf a0 = frag_a(Abuf, KCHUNK, ks);
      v16bf a1 = frag_a(Abuf + 16 * KCHUNK, KCHUNK, ks);
      v16bf b0 = frag_b(Bbuf + 0  * KCHUNK, KCHUNK, ks);
      v16bf b1 = frag_b(Bbuf + 16 * KCHUNK, KCHUNK, ks);
      v16bf b2 = frag_b(Bbuf + 32 * KCHUNK, KCHUNK, ks);
      v16bf b3 = frag_b(Bbuf + 48 * KCHUNK, KCHUNK, ks);
      acc[0][0] = wmma_bf16(a0, b0, acc[0][0]);
      acc[0][1] = wmma_bf16(a0, b1, acc[0][1]);
      acc[0][2] = wmma_bf16(a0, b2, acc[0][2]);
      acc[0][3] = wmma_bf16(a0, b3, acc[0][3]);
      acc[1][0] = wmma_bf16(a1, b0, acc[1][0]);
      acc[1][1] = wmma_bf16(a1, b1, acc[1][1]);
      acc[1][2] = wmma_bf16(a1, b2, acc[1][2]);
      acc[1][3] = wmma_bf16(a1, b3, acc[1][3]);
    }
    __syncthreads();                  // reads done before buffer reuse
  }

  // Epilogue: C/D f32 layout -> VGPR i: lanes0-15 M=i, lanes16-31 M=8+i.
  int lane = threadIdx.x & 31, half = lane >> 4, nn = lane & 15;
  int mbase = mblk + wm * 32;
  int nbase = nblk + wn * 64;
  for (int mt = 0; mt < 2; ++mt)
    for (int nt = 0; nt < 4; ++nt) {
      int col = nbase + nt * 16 + nn;
      float bv = bias[col];
      for (int i = 0; i < 8; ++i) {
        int row = mbase + mt * 16 + half * 8 + i;
        C[(size_t)row * N + col] = acc[mt][nt][i] + bv;
      }
    }
}

// -------------------- persistent recurrent scan -----------------------------
// 16 WGs x 8 waves (latency-bound: direct L2-resident loads, no LDS staging).
// WG g owns output columns [g*64, g*64+64). h ping-pongs in global bf16;
// device-scope barrier = release atomicAdd + acquire spin per step.
__global__ void __launch_bounds__(256)
rnn_scan(const float*  __restrict__ xh,     // [SEQ*BATCH][1024] (incl. b_xh)
         const __bf16* __restrict__ Whh,    // [1024][1024] bf16
         const float*  __restrict__ b_hh,   // [1024]
         __bf16* __restrict__ h_glob,       // [2][BATCH][1024]
         __bf16* __restrict__ hs,           // [SEQ*BATCH][1024]
         unsigned int* __restrict__ counter) {
  const int K = DIM;
  int wid  = threadIdx.x >> 5;
  int lane = threadIdx.x & 31, half = lane >> 4, nn = lane & 15;
  int mt   = wid & 3;                 // batch-row tile: rows mt*16..mt*16+15
  int ntp  = wid >> 2;                // 0..1 -> 32-column half of WG slice
  int cb0  = blockIdx.x * 64 + ntp * 32;

  for (int t = 0; t < SEQ; ++t) {
    const __bf16* hc = h_glob + (size_t)(t & 1) * BATCH * DIM;
    __bf16*       hn = h_glob + (size_t)((t + 1) & 1) * BATCH * DIM;

    v8f acc[2] = {};
    const __bf16* arow = hc + (size_t)(mt * 16) * K;
    for (int k0 = 0; k0 < K; k0 += 32) {
      v16bf a  = frag_a_g(arow, K, k0);
      v16bf b0 = frag_b_g(Whh + (size_t)cb0 * K,        K, k0);
      v16bf b1 = frag_b_g(Whh + (size_t)(cb0 + 16) * K, K, k0);
      acc[0] = wmma_bf16(a, b0, acc[0]);
      acc[1] = wmma_bf16(a, b1, acc[1]);
    }

    for (int j = 0; j < 2; ++j) {
      int col = cb0 + j * 16 + nn;
      float bh = b_hh[col];
      for (int i = 0; i < 8; ++i) {
        int b = mt * 16 + half * 8 + i;
        size_t ro = (size_t)(t * BATCH + b) * DIM + col;
        float v = tanhf(acc[j][i] + xh[ro] + bh);
        __bf16 hv = (__bf16)v;
        hn[(size_t)b * DIM + col] = hv;   // next-step h
        hs[ro] = hv;                      // saved for output GEMM
      }
    }

    // ---- device-scope grid barrier (arrive + wait) ----
    __threadfence();
    __syncthreads();
    if (threadIdx.x == 0) {
      __hip_atomic_fetch_add(counter, 1u, __ATOMIC_RELEASE,
                             __HIP_MEMORY_SCOPE_AGENT);
      unsigned target = (unsigned)(t + 1) * SCAN_WGS;
      while (__hip_atomic_load(counter, __ATOMIC_ACQUIRE,
                               __HIP_MEMORY_SCOPE_AGENT) < target) {
        __builtin_amdgcn_s_sleep(1);
      }
    }
    __syncthreads();
  }
}

// -------------------- launcher ----------------------------------------------
extern "C" void kernel_launch(void* const* d_in, const int* in_sizes, int n_in,
                              void* d_out, int out_size, void* d_ws,
                              size_t ws_size, hipStream_t stream) {
  const float* x    = (const float*)d_in[0];
  const float* W_xh = (const float*)d_in[1];
  const float* b_xh = (const float*)d_in[2];
  const float* W_hh = (const float*)d_in[3];
  const float* b_hh = (const float*)d_in[4];
  const float* W_hy = (const float*)d_in[5];
  const float* b_hy = (const float*)d_in[6];
  float* out = (float*)d_out;

  char* ws = (char*)d_ws;                    // requires ~263 MiB scratch
  __bf16* Wxh_b = (__bf16*)(ws + OFF_WXH);
  __bf16* Whh_b = (__bf16*)(ws + OFF_WHH);
  __bf16* Why_b = (__bf16*)(ws + OFF_WHY);
  __bf16* x_b   = (__bf16*)(ws + OFF_XB);
  float*  xh    = (float*) (ws + OFF_XH);
  __bf16* hs_b  = (__bf16*)(ws + OFF_HS);
  __bf16* h_gl  = (__bf16*)(ws + OFF_HG);
  unsigned int* ctr = (unsigned int*)(ws + OFF_CTR);

  // 1) precision conversion (weights + x) into workspace
  f32_to_bf16<<<256,  256, 0, stream>>>(W_xh, Wxh_b, (int)SZ_W);
  f32_to_bf16<<<256,  256, 0, stream>>>(W_hh, Whh_b, (int)SZ_W);
  f32_to_bf16<<<256,  256, 0, stream>>>(W_hy, Why_b, (int)SZ_W);
  f32_to_bf16<<<2048, 256, 0, stream>>>(x,    x_b,   (int)SZ_X);

  // 2) xh = x @ W_xh^T + b_xh   (32768 x 1024 x 1024, TDM-staged)
  dim3 gg(MTOT / 64, DIM / 256);
  gemm_bf16_bias<<<gg, 256, LDS_BYTES, stream>>>(x_b, Wxh_b, b_xh, xh, MTOT);

  // 3) recurrent scan (h0 = 0, barrier counter = 0; reset every call so
  //    graph replays are deterministic)
  (void)hipMemsetAsync(h_gl, 0, (size_t)BATCH * DIM * 2, stream);
  (void)hipMemsetAsync(ctr,  0, sizeof(unsigned int), stream);
  rnn_scan<<<SCAN_WGS, 256, 0, stream>>>(xh, Whh_b, b_hh, h_gl, hs_b, ctr);

  // 4) out = hs @ W_hy^T + b_hy (TDM-staged)
  gemm_bf16_bias<<<gg, 256, LDS_BYTES, stream>>>(hs_b, Why_b, b_hy, out, MTOT);
}
